// ChampionNet_33784212750970
// MI455X (gfx1250) — compile-verified, matrix operand order checked by reading, not run on hardware
//
#include <hip/hip_runtime.h>
#include <hip/hip_bf16.h>

typedef _Float16 h16;
typedef __attribute__((ext_vector_type(16))) _Float16 v16h;
typedef __attribute__((ext_vector_type(8)))  _Float16 v8h;
typedef __attribute__((ext_vector_type(8)))  float    v8f;

typedef unsigned int u32x4 __attribute__((ext_vector_type(4)));
typedef int          i32x4 __attribute__((ext_vector_type(4)));
typedef int          i32x8 __attribute__((ext_vector_type(8)));

#if defined(__has_builtin)
#  if __has_builtin(__builtin_amdgcn_tensor_load_to_lds) && \
      __has_builtin(__builtin_amdgcn_s_wait_tensorcnt)
#    define HAVE_TDM 1
#  endif
#endif
#ifndef HAVE_TDM
#  define HAVE_TDM 0
#endif

#define DEVI static __device__ __forceinline__

// Load a 16-wide f16 fragment in the CDNA5 WMMA A/B register layout.
// p points at element k = (hh*8) of this lane's row; elements 0..7 <- k..k+7,
// elements 8..15 <- k+16..k+23 (two b128 loads).
DEVI v16h load_frag(const h16* p) {
  v16h r;
  v8h lo = *(const v8h*)p;
  v8h hi = *(const v8h*)(p + 16);
#pragma unroll
  for (int i = 0; i < 8; ++i) { r[i] = lo[i]; r[8 + i] = hi[i]; }
  return r;
}

DEVI v8f wmma16(v16h a, v16h b, v8f c) {
  return __builtin_amdgcn_wmma_f32_16x16x32_f16(false, a, false, b, (short)0, c,
                                                false, false);
}

// ---------------------------------------------------------------------------
// TDM warm kernel: stage a 64x16 f16 tile of a weight matrix into LDS via the
// Tensor Data Mover (async DMA, TENSORcnt), then discard.  Side effect: pulls
// the first weight lines into L2 ahead of the GEMMs.  `never` == 0 at runtime.
// ---------------------------------------------------------------------------
__global__ __launch_bounds__(32) void tdm_warm_kernel(const h16* __restrict__ src,
                                                      float* __restrict__ sink,
                                                      int never) {
#if HAVE_TDM
  __shared__ __align__(16) h16 stage[2048];  // 4 KB LDS staging buffer
  unsigned long long ga = (unsigned long long)(const void*)src;
  // D# group 0: count=1 (valid), lds_addr=0, global_addr, type=2 ("image").
  u32x4 g0 = {1u, 0u, (unsigned)(ga & 0xffffffffu),
              (unsigned)((ga >> 32) & 0x01ffffffu) | 0x80000000u};
  // D# group 1: data_size=1 (2B), tensor_dim0=64, tensor_dim1=64,
  // tile_dim0=64, tile_dim1=16, tensor_dim0_stride=64.
  i32x8 g1 = {1 << 16,            // data_size = 2 bytes
              64 << 16,           // tensor_dim0[15:0]
              64 << 16,           // tensor_dim0 hi | tensor_dim1[15:0]
              64 << 16,           // tensor_dim1 hi | tile_dim0 = 64
              16,                 // tile_dim1 = 16, tile_dim2 = 0
              64,                 // tensor_dim0_stride lo
              0, 0};
  i32x4 gz = {0, 0, 0, 0};
#if __clang_major__ >= 23
  i32x8 gz8 = {0, 0, 0, 0, 0, 0, 0, 0};
  __builtin_amdgcn_tensor_load_to_lds(g0, g1, gz, gz, gz8, 0);
#else
  __builtin_amdgcn_tensor_load_to_lds(g0, g1, gz, gz, 0);
#endif
  __builtin_amdgcn_s_wait_tensorcnt(0);
  __syncthreads();
  if (never) sink[0] = (float)stage[never & 2047];  // keep LDS + DMA observable
#else
  (void)src; (void)sink; (void)never;
#endif
}

// ---------------------------------------------------------------------------
// Generic GEMM: out[M,N] = A[M,K] * W[N,K]^T (+ bias[N]).  K % 32 == 0,
// M % 32 == 0.  Block = 128 threads (4 waves); wave w computes a 32x64 tile:
// rows [bx*32, +32) x cols [(by*4+w)*64, +64), 8 WMMAs per K-step.
// Out-of-range B rows are clamped to N-1 (their product columns are never
// stored), so the inner loop has no exec-mask manipulation at all.
// ---------------------------------------------------------------------------
__global__ __launch_bounds__(128) void gemm_wmma_kernel(
    const h16* __restrict__ A, const h16* __restrict__ W,
    const float* __restrict__ bias, float* __restrict__ out,
    int M, int N, int K) {
  const int wave = threadIdx.x >> 5;
  const int lane = threadIdx.x & 31;
  const int m0 = blockIdx.x * 32;
  const int n0 = (blockIdx.y * 4 + wave) * 64;
  if (n0 >= N || m0 >= M) return;
  const int mr = lane & 15;
  const int hh = lane >> 4;
  const h16* Ar0 = A + (size_t)(m0 + mr) * K + hh * 8;
  const h16* Ar1 = A + (size_t)(m0 + 16 + mr) * K + hh * 8;
  const int Nm1 = N - 1;
  const int na = n0 + mr, nb = n0 + 16 + mr, nc = n0 + 32 + mr, nd = n0 + 48 + mr;
  const h16* Wa = W + (size_t)(na < Nm1 ? na : Nm1) * K + hh * 8;
  const h16* Wb = W + (size_t)(nb < Nm1 ? nb : Nm1) * K + hh * 8;
  const h16* Wc = W + (size_t)(nc < Nm1 ? nc : Nm1) * K + hh * 8;
  const h16* Wd = W + (size_t)(nd < Nm1 ? nd : Nm1) * K + hh * 8;
  v8f c00 = {}, c01 = {}, c02 = {}, c03 = {};
  v8f c10 = {}, c11 = {}, c12 = {}, c13 = {};
  for (int kk = 0; kk < K; kk += 32) {
    if (kk + 32 < K) {
      __builtin_prefetch(Ar0 + kk + 32, 0, 1);
      __builtin_prefetch(Ar1 + kk + 32, 0, 1);
    }
    v16h a0 = load_frag(Ar0 + kk);
    v16h a1 = load_frag(Ar1 + kk);
    v16h b0 = load_frag(Wa + kk);
    v16h b1 = load_frag(Wb + kk);
    v16h b2 = load_frag(Wc + kk);
    v16h b3 = load_frag(Wd + kk);
    c00 = wmma16(a0, b0, c00);
    c10 = wmma16(a1, b0, c10);
    c01 = wmma16(a0, b1, c01);
    c11 = wmma16(a1, b1, c11);
    c02 = wmma16(a0, b2, c02);
    c12 = wmma16(a1, b2, c12);
    c03 = wmma16(a0, b3, c03);
    c13 = wmma16(a1, b3, c13);
  }
  const float ba = (bias && na < N) ? bias[na] : 0.f;
  const float bb = (bias && nb < N) ? bias[nb] : 0.f;
  const float bc = (bias && nc < N) ? bias[nc] : 0.f;
  const float bd = (bias && nd < N) ? bias[nd] : 0.f;
#pragma unroll
  for (int r = 0; r < 8; ++r) {
    size_t r0 = (size_t)(m0 + 8 * hh + r) * N;
    size_t r1 = (size_t)(m0 + 16 + 8 * hh + r) * N;
    if (na < N) { out[r0 + na] = c00[r] + ba; out[r1 + na] = c10[r] + ba; }
    if (nb < N) { out[r0 + nb] = c01[r] + bb; out[r1 + nb] = c11[r] + bb; }
    if (nc < N) { out[r0 + nc] = c02[r] + bc; out[r1 + nc] = c12[r] + bc; }
    if (nd < N) { out[r0 + nd] = c03[r] + bd; out[r1 + nd] = c13[r] + bd; }
  }
}

// ---------------------------------------------------------------------------
// Fused attention for Dh=32 heads: one wave per 16 query rows of one
// (slice) = (group, head).  Q pre-scaled by 1/sqrt(32).  Online softmax with
// shfl_xor reductions inside the 16-lane half; P goes through LDS to be
// re-laid-out as an A-fragment for the P*V WMMAs.  V is pre-transposed
// (Vt[slice][d][t]).  Output row = (s/H)*T + q, col = (s%H)*32 + d.
// ---------------------------------------------------------------------------
__global__ __launch_bounds__(32) void attn_wmma_kernel(
    const h16* __restrict__ Q, const h16* __restrict__ Kh,
    const h16* __restrict__ Vt, float* __restrict__ Out,
    int T, int H, int outLd) {
  const int s = blockIdx.y;
  const int q0 = blockIdx.x * 16;
  const int lane = threadIdx.x;
  const int mr = lane & 15;
  const int hh = lane >> 4;
  __shared__ __align__(16) h16 Pt[16][32];

  v16h qf = load_frag(Q + ((size_t)s * T + q0 + mr) * 32 + hh * 8);
  const h16* Ks = Kh + (size_t)s * T * 32;
  const h16* Vs = Vt + (size_t)s * 32 * T;

  v8f o0 = {}, o1 = {};
  float rowM[8], rowL[8];
#pragma unroll
  for (int r = 0; r < 8; ++r) { rowM[r] = -1e30f; rowL[r] = 0.f; }

  for (int kt = 0; kt < T; kt += 32) {
    v16h kb0 = load_frag(Ks + (size_t)(kt + mr) * 32 + hh * 8);
    v16h kb1 = load_frag(Ks + (size_t)(kt + 16 + mr) * 32 + hh * 8);
    v8f st0 = {}, st1 = {};
    st0 = wmma16(qf, kb0, st0);
    st1 = wmma16(qf, kb1, st1);
#pragma unroll
    for (int r = 0; r < 8; ++r) {
      float v0 = st0[r], v1 = st1[r];
      float mx = fmaxf(v0, v1);
      for (int off = 1; off < 16; off <<= 1)
        mx = fmaxf(mx, __shfl_xor(mx, off, 32));
      float mn = fmaxf(rowM[r], mx);
      float corr = __expf(rowM[r] - mn);
      float p0 = __expf(v0 - mn);
      float p1 = __expf(v1 - mn);
      float ps = p0 + p1;
      for (int off = 1; off < 16; off <<= 1) ps += __shfl_xor(ps, off, 32);
      rowL[r] = rowL[r] * corr + ps;
      rowM[r] = mn;
      o0[r] *= corr;
      o1[r] *= corr;
      int m = r + 8 * hh;
      Pt[m][mr] = (h16)p0;
      Pt[m][16 + mr] = (h16)p1;
    }
    __syncthreads();
    v16h pf = load_frag(&Pt[mr][hh * 8]);
    __syncthreads();
    v16h vb0 = load_frag(Vs + (size_t)mr * T + kt + hh * 8);
    v16h vb1 = load_frag(Vs + (size_t)(16 + mr) * T + kt + hh * 8);
    o0 = wmma16(pf, vb0, o0);
    o1 = wmma16(pf, vb1, o1);
  }

  size_t rowbase = (size_t)(s / H) * T + q0;
  int colbase = (s % H) * 32;
#pragma unroll
  for (int r = 0; r < 8; ++r) {
    int m = r + 8 * hh;
    float inv = 1.0f / rowL[r];
    size_t row = (rowbase + m) * (size_t)outLd;
    Out[row + colbase + mr] = o0[r] * inv;
    Out[row + colbase + 16 + mr] = o1[r] * inv;
  }
}

// ---------------------------------------------------------------------------
// LayerNorm (no bias): one block per row, D threads (power of two).
// ---------------------------------------------------------------------------
__global__ void ln_kernel(const float* __restrict__ src, int ldSrc, int colOff,
                          const float* __restrict__ w, float* __restrict__ out32,
                          h16* __restrict__ out16, int D) {
  extern __shared__ float red[];
  int row = blockIdx.x, tid = threadIdx.x;
  float v = src[(size_t)row * ldSrc + colOff + tid];
  red[tid] = v;
  __syncthreads();
  for (int s = D >> 1; s > 0; s >>= 1) {
    if (tid < s) red[tid] += red[tid + s];
    __syncthreads();
  }
  float mean = red[0] / D;
  __syncthreads();
  float dv = v - mean;
  red[tid] = dv * dv;
  __syncthreads();
  for (int s = D >> 1; s > 0; s >>= 1) {
    if (tid < s) red[tid] += red[tid + s];
    __syncthreads();
  }
  float var = red[0] / D;
  float y = dv * rsqrtf(var + 1e-5f) * w[tid];
  if (out32) out32[(size_t)row * D + tid] = y;
  if (out16) out16[(size_t)row * D + tid] = (h16)y;
}

// ---------------------------------------------------------------------------
// Elementwise / packing helpers
// ---------------------------------------------------------------------------
__global__ void convert_f32_f16_kernel(const float* __restrict__ a,
                                       h16* __restrict__ b, long long n) {
  long long i = (long long)blockIdx.x * blockDim.x + threadIdx.x;
  if (i < n) b[i] = (h16)a[i];
}

__global__ void resadd_kernel(float* __restrict__ x, const float* __restrict__ y,
                              const float* __restrict__ gamma, int C, long long n) {
  long long i = (long long)blockIdx.x * blockDim.x + threadIdx.x;
  if (i >= n) return;
  float g = gamma ? gamma[i % C] : 1.f;
  x[i] += y[i] * g;
}

// src[B,T,H*32] (ld=ldSrc) -> head-sliced f16: dst[(b*H+h)*T + t][d]
// (or transposed dst[(b*H+h)*32 + d][t] when tr != 0), scaled.
__global__ void pack_heads_kernel(const float* __restrict__ src, int ldSrc,
                                  h16* __restrict__ dst, int T, int H,
                                  float scale, int tr, int n) {
  int gid = blockIdx.x * blockDim.x + threadIdx.x;
  if (gid >= n) return;
  int d = gid & 31;
  int r = gid >> 5;
  int h = r % H; r /= H;
  int t = r % T;
  int bb = r / T;
  float v = src[((size_t)bb * T + t) * ldSrc + h * 32 + d] * scale;
  size_t slice = (size_t)bb * H + h;
  if (!tr) dst[(slice * T + t) * 32 + d] = (h16)v;
  else     dst[(slice * 32 + d) * (size_t)T + t] = (h16)v;
}

// mamba: u = (causal depthwise conv4 of xz[...,:512] + cb) * sigmoid(xz[...,512:])
__global__ void conv_gate_kernel(const float* __restrict__ xz,
                                 const float* __restrict__ cw,
                                 const float* __restrict__ cb,
                                 float* __restrict__ u, h16* __restrict__ u16,
                                 int T, int n) {
  int gid = blockIdx.x * blockDim.x + threadIdx.x;
  if (gid >= n) return;
  int d = gid % 512;
  int row = gid / 512;
  int t = row % T;
  int b = row / T;
  float acc = cb[d];
#pragma unroll
  for (int j = 0; j < 4; ++j) {
    int tt = t - 3 + j;
    if (tt >= 0) acc += cw[d * 4 + j] * xz[((size_t)b * T + tt) * 1024 + d];
  }
  float z = xz[((size_t)b * T + t) * 1024 + 512 + d];
  float uu = acc * (1.f / (1.f + __expf(-z)));
  u[gid] = uu;
  u16[gid] = (h16)uu;
}

__global__ void dt_kernel(const float* __restrict__ xp, const float* __restrict__ dtw,
                          const float* __restrict__ dtb, float* __restrict__ dt, int n) {
  int gid = blockIdx.x * blockDim.x + threadIdx.x;
  if (gid >= n) return;
  int d = gid % 512;
  int row = gid / 512;
  float v = xp[(size_t)row * 33] * dtw[d] + dtb[d];
  dt[gid] = (v > 20.f) ? v : log1pf(__expf(v));
}

// Sequential SSM scan: one thread per (b, d) channel, 16 states in registers.
__global__ void scan_kernel(const float* __restrict__ dt, const float* __restrict__ u,
                            const float* __restrict__ xp, const float* __restrict__ A,
                            const float* __restrict__ Dp, float* __restrict__ ys,
                            int B, int T) {
  int id = blockIdx.x * blockDim.x + threadIdx.x;
  if (id >= B * 512) return;
  int b = id / 512, d = id % 512;
  float a[16], st[16];
#pragma unroll
  for (int i = 0; i < 16; ++i) { a[i] = A[d * 16 + i]; st[i] = 0.f; }
  float Dd = Dp[d];
  for (int t = 0; t < T; ++t) {
    size_t row = (size_t)b * T + t;
    float dtv = dt[row * 512 + d];
    float uv = u[row * 512 + d];
    float du = dtv * uv;
    const float* xr = xp + row * 33;
    float y = 0.f;
#pragma unroll
    for (int i = 0; i < 16; ++i) {
      float dA = __expf(dtv * a[i]);
      st[i] = st[i] * dA + du * xr[1 + i];
      y += st[i] * xr[17 + i];
    }
    ys[row * 512 + d] = y + Dd * uv;
  }
}

// mamba tail: x += (depthwise conv7, pad 3/3, of yo) * gamma
__global__ void postconv_res_kernel(float* __restrict__ x, const float* __restrict__ yo,
                                    const float* __restrict__ pw,
                                    const float* __restrict__ gamma, int T, int n) {
  int gid = blockIdx.x * blockDim.x + threadIdx.x;
  if (gid >= n) return;
  int c = gid % 256;
  int row = gid / 256;
  int t = row % T;
  int b = row / T;
  float acc = 0.f;
#pragma unroll
  for (int j = 0; j < 7; ++j) {
    int tt = t - 3 + j;
    if (tt >= 0 && tt < T) acc += pw[c * 7 + j] * yo[((size_t)b * T + tt) * 256 + c];
  }
  x[gid] += acc * gamma[c];
}

__global__ void router_kernel(const float* __restrict__ xn, const float* __restrict__ rw,
                              const float* __restrict__ rb, float* __restrict__ gate,
                              int rows) {
  int row = blockIdx.x * blockDim.x + threadIdx.x;
  if (row >= rows) return;
  const float* xr = xn + (size_t)row * 256;
  float lg[4];
  float mx = -1e30f;
#pragma unroll
  for (int e = 0; e < 4; ++e) {
    float a = rb[e];
    const float* w = rw + e * 256;
    for (int j = 0; j < 256; ++j) a += xr[j] * w[j];
    lg[e] = a;
    mx = fmaxf(mx, a);
  }
  float s = 0.f;
#pragma unroll
  for (int e = 0; e < 4; ++e) { lg[e] = __expf(lg[e] - mx); s += lg[e]; }
#pragma unroll
  for (int e = 0; e < 4; ++e) gate[(size_t)row * 4 + e] = lg[e] / s;
}

__global__ void pack_chunks_kernel(const float* __restrict__ xn, h16* __restrict__ Xc,
                                   int T, int n) {
  int gid = blockIdx.x * blockDim.x + threadIdx.x;
  if (gid >= n) return;
  int col = gid % 256;
  int row = gid / 256;
  int t = row % T;
  int b = row / T;
  int c = col >> 6, j = col & 63;
  Xc[(((size_t)(b * 4 + c) * T + t) << 6) + j] = (h16)xn[gid];
}

__global__ void gate_accum_kernel(float* __restrict__ acc, const float* __restrict__ Ye,
                                  const float* __restrict__ gate, int T, int ei, int n) {
  int gid = blockIdx.x * blockDim.x + threadIdx.x;
  if (gid >= n) return;
  int bcrow = gid >> 6;
  int t = bcrow % T;
  int b = (bcrow / T) >> 2;
  float g = gate[((size_t)b * T + t) * 4 + ei];
  float v = g * Ye[gid];
  acc[gid] = (ei == 0) ? v : acc[gid] + v;
}

__global__ void unpack_acc_kernel(const float* __restrict__ acc, h16* __restrict__ ym,
                                  int T, int n) {
  int gid = blockIdx.x * blockDim.x + threadIdx.x;
  if (gid >= n) return;
  int col = gid % 256;
  int row = gid / 256;
  int t = row % T;
  int b = row / T;
  int c = col >> 6, j = col & 63;
  ym[gid] = (h16)acc[(((size_t)(b * 4 + c) * T + t) << 6) + j];
}

__global__ void add_strided_kernel(float* __restrict__ x, int ldX, int off,
                                   const float* __restrict__ t, int D, int n) {
  int gid = blockIdx.x * blockDim.x + threadIdx.x;
  if (gid >= n) return;
  int c = gid % D;
  int row = gid / D;
  x[(size_t)row * ldX + off + c] += t[gid];
}

__global__ void silu_f16_kernel(const float* __restrict__ a, h16* __restrict__ b,
                                int n) {
  int i = blockIdx.x * blockDim.x + threadIdx.x;
  if (i >= n) return;
  float v = a[i];
  b[i] = (h16)(v * (1.f / (1.f + __expf(-v))));
}

__global__ void sigmoid_kernel(const float* __restrict__ a, float* __restrict__ b,
                               int n) {
  int i = blockIdx.x * blockDim.x + threadIdx.x;
  if (i >= n) return;
  b[i] = 1.f / (1.f + __expf(-a[i]));
}

// saliency: rank-count top-k mask, fused with residual; writes final output.
__global__ void topk_residual_kernel(const float* __restrict__ xcur,
                                     const float* __restrict__ xn,
                                     const float* __restrict__ s,
                                     const float* __restrict__ gamma,
                                     float* __restrict__ out, int Ktop) {
  __shared__ float sv[256];
  int row = blockIdx.x, i = threadIdx.x;
  float si = s[(size_t)row * 256 + i];
  sv[i] = si;
  __syncthreads();
  int cnt = 0;
  for (int j = 0; j < 256; ++j) {
    float sj = sv[j];
    cnt += (sj > si) || (sj == si && j < i);
  }
  float add = (cnt < Ktop) ? xn[(size_t)row * 256 + i] * gamma[i] : 0.f;
  out[(size_t)row * 256 + i] = xcur[(size_t)row * 256 + i] + add;
}

// ---------------------------------------------------------------------------
// Host orchestration
// ---------------------------------------------------------------------------
extern "C" void kernel_launch(void* const* d_in, const int* in_sizes, int n_in,
                              void* d_out, int out_size, void* d_ws, size_t ws_size,
                              hipStream_t stream) {
  (void)in_sizes; (void)n_in; (void)out_size; (void)ws_size;
  constexpr int B = 4, T = 2048, C = 256;
  constexpr int NR = B * T;                 // 8192 rows
  constexpr float INVS32 = 0.17677669529663688f;  // 1/sqrt(32)

  const float* X        = (const float*)d_in[0];
  const float* mh_norm  = (const float*)d_in[1];
  const float* mh_q0    = (const float*)d_in[2];
  const float* mh_qb    = (const float*)d_in[3];
  const float* mh_q2    = (const float*)d_in[4];
  const float* mh_kv0   = (const float*)d_in[5];
  const float* mh_kvb   = (const float*)d_in[6];
  const float* mh_k     = (const float*)d_in[7];
  const float* mh_v     = (const float*)d_in[8];
  const float* mh_o     = (const float*)d_in[9];
  const float* mh_gamma = (const float*)d_in[10];
  const float* mb_norm  = (const float*)d_in[11];
  const float* mb_inpr  = (const float*)d_in[12];
  const float* mb_cw    = (const float*)d_in[13];
  const float* mb_cb    = (const float*)d_in[14];
  const float* mb_xproj = (const float*)d_in[15];
  const float* mb_dtw   = (const float*)d_in[16];
  const float* mb_dtb   = (const float*)d_in[17];
  const float* mb_A     = (const float*)d_in[18];
  const float* mb_D     = (const float*)d_in[19];
  const float* mb_outpr = (const float*)d_in[20];
  const float* mb_postw = (const float*)d_in[21];
  const float* mb_gamma = (const float*)d_in[22];
  const float* mo_norm  = (const float*)d_in[23];
  const float* mo_rw    = (const float*)d_in[24];
  const float* mo_rb    = (const float*)d_in[25];
  const float* mo_qkv   = (const float*)d_in[26];
  const float* mo_projw = (const float*)d_in[27];
  const float* mo_projb = (const float*)d_in[28];
  const float* mo_outw  = (const float*)d_in[29];
  const float* rv_fn    = (const float*)d_in[30];
  const float* rv_fw    = (const float*)d_in[31];
  const float* rv_gn    = (const float*)d_in[32];
  const float* rv_gw    = (const float*)d_in[33];
  const float* sl_norm  = (const float*)d_in[34];
  const float* sl_w1    = (const float*)d_in[35];
  const float* sl_b1    = (const float*)d_in[36];
  const float* sl_w2    = (const float*)d_in[37];
  const float* sl_b2    = (const float*)d_in[38];
  const float* sl_gamma = (const float*)d_in[39];

  char* base = (char*)d_ws;
  size_t woff = 0;
  auto walloc = [&](size_t bytes) -> char* {
    char* p = base + woff;
    woff = (woff + bytes + 255) & ~(size_t)255;
    return p;
  };
  auto ew = [](long long n) { return dim3((unsigned)((n + 255) / 256)); };
  auto cvt_to = [&](const float* src, h16* dst, long long n) {
    convert_f32_f16_kernel<<<ew(n), 256, 0, stream>>>(src, dst, n);
  };
  auto cvt = [&](const float* src, long long n) -> h16* {
    h16* dst = (h16*)walloc((size_t)n * sizeof(h16));
    cvt_to(src, dst, n);
    return dst;
  };
  auto gemm = [&](const h16* A, const h16* W, const float* bias, float* out,
                  int M, int N, int K) {
    dim3 g(M / 32, (N + 255) / 256);
    gemm_wmma_kernel<<<g, 128, 0, stream>>>(A, W, bias, out, M, N, K);
  };
  auto ln = [&](const float* src, int ld, int off, const float* w, float* o32,
                h16* o16, int D, int rows) {
    ln_kernel<<<rows, D, D * sizeof(float), stream>>>(src, ld, off, w, o32, o16, D);
  };

  // ---- persistent: f16 weights + residual stream -------------------------
  h16* w_q0    = cvt(mh_q0, 32 * 256);
  h16* w_q2    = cvt(mh_q2, 256 * 32);
  h16* w_kv0   = cvt(mh_kv0, 64 * 256);
  h16* w_k     = cvt(mh_k, 256 * 64);
  h16* w_v     = cvt(mh_v, 256 * 64);
  h16* w_o     = cvt(mh_o, 256 * 256);
  h16* w_inpr  = cvt(mb_inpr, 1024 * 256);
  h16* w_xproj = cvt(mb_xproj, 33 * 512);
  h16* w_outpr = cvt(mb_outpr, 256 * 512);
  h16* w_qkv   = cvt(mo_qkv, 4 * 192 * 64);
  h16* w_projw = cvt(mo_projw, 4 * 64 * 64);
  h16* w_outw  = cvt(mo_outw, 256 * 256);
  h16* w_fw    = cvt(rv_fw, 128 * 128);
  h16* w_gw    = cvt(rv_gw, 128 * 128);
  h16* w_w1    = cvt(sl_w1, 64 * 256);
  h16* w_w2    = cvt(sl_w2, 256 * 64);

  float* xcur = (float*)walloc((size_t)NR * C * sizeof(float));
  hipMemcpyAsync(xcur, X, (size_t)NR * C * sizeof(float),
                 hipMemcpyDeviceToDevice, stream);
  const size_t scratch0 = woff;

  // Async-tensor (TDM) warm of the large weight matrices into L2/LDS.
  tdm_warm_kernel<<<dim3(1), 32, 0, stream>>>(w_inpr, xcur, 0);
  tdm_warm_kernel<<<dim3(1), 32, 0, stream>>>(w_outpr, xcur, 0);
  tdm_warm_kernel<<<dim3(1), 32, 0, stream>>>(w_o, xcur, 0);
  tdm_warm_kernel<<<dim3(1), 32, 0, stream>>>(w_outw, xcur, 0);

  // ======================= Phase 1: MHLA ==================================
  {
    woff = scratch0;
    float* xn = (float*)walloc((size_t)NR * C * 4);
    h16* xn16 = (h16*)walloc((size_t)NR * C * 2);
    ln(xcur, C, 0, mh_norm, xn, xn16, C, NR);

    float* qmid = (float*)walloc((size_t)NR * 32 * 4);
    gemm(xn16, w_q0, mh_qb, qmid, NR, 32, 256);
    h16* qmid16 = (h16*)walloc((size_t)NR * 32 * 2);
    cvt_to(qmid, qmid16, (long long)NR * 32);
    float* qfull = (float*)walloc((size_t)NR * C * 4);
    gemm(qmid16, w_q2, nullptr, qfull, NR, 256, 32);

    float* kvm = (float*)walloc((size_t)NR * 64 * 4);
    gemm(xn16, w_kv0, mh_kvb, kvm, NR, 64, 256);
    h16* kv16 = (h16*)walloc((size_t)NR * 64 * 2);
    cvt_to(kvm, kv16, (long long)NR * 64);
    float* kfull = (float*)walloc((size_t)NR * C * 4);
    float* vfull = (float*)walloc((size_t)NR * C * 4);
    gemm(kv16, w_k, nullptr, kfull, NR, 256, 64);
    gemm(kv16, w_v, nullptr, vfull, NR, 256, 64);

    int np = NR * C;  // elements per pack
    h16* Qh = (h16*)walloc((size_t)np * 2);
    h16* Kh = (h16*)walloc((size_t)np * 2);
    h16* Vt = (h16*)walloc((size_t)np * 2);
    pack_heads_kernel<<<ew(np), 256, 0, stream>>>(qfull, 256, Qh, T, 8, INVS32, 0, np);
    pack_heads_kernel<<<ew(np), 256, 0, stream>>>(kfull, 256, Kh, T, 8, 1.f, 0, np);
    pack_heads_kernel<<<ew(np), 256, 0, stream>>>(vfull, 256, Vt, T, 8, 1.f, 1, np);

    float* yat = (float*)walloc((size_t)NR * C * 4);
    attn_wmma_kernel<<<dim3(T / 16, B * 8), 32, 0, stream>>>(Qh, Kh, Vt, yat, T, 8, 256);
    h16* yat16 = (h16*)walloc((size_t)NR * C * 2);
    cvt_to(yat, yat16, (long long)NR * C);
    float* oout = (float*)walloc((size_t)NR * C * 4);
    gemm(yat16, w_o, nullptr, oout, NR, 256, 256);
    resadd_kernel<<<ew((long long)NR * C), 256, 0, stream>>>(xcur, oout, mh_gamma, C,
                                                             (long long)NR * C);
  }

  // ======================= Phase 2: Mamba =================================
  {
    woff = scratch0;
    float* xn = (float*)walloc((size_t)NR * C * 4);
    h16* xn16 = (h16*)walloc((size_t)NR * C * 2);
    ln(xcur, C, 0, mb_norm, xn, xn16, C, NR);

    float* xz = (float*)walloc((size_t)NR * 1024 * 4);
    gemm(xn16, w_inpr, nullptr, xz, NR, 1024, 256);

    float* u = (float*)walloc((size_t)NR * 512 * 4);
    h16* u16 = (h16*)walloc((size_t)NR * 512 * 2);
    conv_gate_kernel<<<ew((long long)NR * 512), 256, 0, stream>>>(
        xz, mb_cw, mb_cb, u, u16, T, NR * 512);

    float* xp = (float*)walloc((size_t)NR * 33 * 4);
    gemm(u16, w_xproj, nullptr, xp, NR, 33, 512);

    float* dt = (float*)walloc((size_t)NR * 512 * 4);
    dt_kernel<<<ew((long long)NR * 512), 256, 0, stream>>>(xp, mb_dtw, mb_dtb, dt,
                                                           NR * 512);
    float* ys = (float*)walloc((size_t)NR * 512 * 4);
    scan_kernel<<<dim3((B * 512 + 255) / 256), 256, 0, stream>>>(dt, u, xp, mb_A,
                                                                 mb_D, ys, B, T);
    h16* ys16 = (h16*)walloc((size_t)NR * 512 * 2);
    cvt_to(ys, ys16, (long long)NR * 512);
    float* yo = (float*)walloc((size_t)NR * C * 4);
    gemm(ys16, w_outpr, nullptr, yo, NR, 256, 512);
    postconv_res_kernel<<<ew((long long)NR * C), 256, 0, stream>>>(
        xcur, yo, mb_postw, mb_gamma, T, NR * C);
  }

  // ======================= Phase 3: MoA ===================================
  {
    woff = scratch0;
    float* xn = (float*)walloc((size_t)NR * C * 4);
    h16* xn16 = (h16*)walloc((size_t)NR * C * 2);
    ln(xcur, C, 0, mo_norm, xn, xn16, C, NR);

    float* gate = (float*)walloc((size_t)NR * 4 * 4);
    router_kernel<<<ew(NR), 256, 0, stream>>>(xn, mo_rw, mo_rb, gate, NR);

    const int MR = NR * 4;  // 32768 chunk-rows (b,c,t)
    h16* Xc = (h16*)walloc((size_t)MR * 64 * 2);
    pack_chunks_kernel<<<ew((long long)NR * C), 256, 0, stream>>>(xn, Xc, T, NR * C);

    float* QKVe = (float*)walloc((size_t)MR * 192 * 4);
    h16* Qh = (h16*)walloc((size_t)MR * 64 * 2);
    h16* Kh = (h16*)walloc((size_t)MR * 64 * 2);
    h16* Vt = (h16*)walloc((size_t)MR * 64 * 2);
    float* Oe = (float*)walloc((size_t)MR * 64 * 4);
    h16* Oe16 = (h16*)walloc((size_t)MR * 64 * 2);
    float* Ye = (float*)walloc((size_t)MR * 64 * 4);
    float* accb = (float*)walloc((size_t)MR * 64 * 4);

    const int np = MR * 64;  // 2.1M pack elements (16 groups x T x 2 heads x 32)
    for (int ei = 0; ei < 4; ++ei) {
      gemm(Xc, w_qkv + (size_t)ei * 192 * 64, nullptr, QKVe, MR, 192, 64);
      pack_heads_kernel<<<ew(np), 256, 0, stream>>>(QKVe, 192, Qh, T, 2, INVS32, 0, np);
      pack_heads_kernel<<<ew(np), 256, 0, stream>>>(QKVe + 64, 192, Kh, T, 2, 1.f, 0, np);
      pack_heads_kernel<<<ew(np), 256, 0, stream>>>(QKVe + 128, 192, Vt, T, 2, 1.f, 1, np);
      attn_wmma_kernel<<<dim3(T / 16, 32), 32, 0, stream>>>(Qh, Kh, Vt, Oe, T, 2, 64);
      cvt_to(Oe, Oe16, (long long)MR * 64);
      gemm(Oe16, w_projw + (size_t)ei * 64 * 64, mo_projb + (size_t)ei * 64, Ye,
           MR, 64, 64);
      gate_accum_kernel<<<ew((long long)MR * 64), 256, 0, stream>>>(accb, Ye, gate, T,
                                                                    ei, MR * 64);
    }
    h16* ym16 = (h16*)walloc((size_t)NR * C * 2);
    unpack_acc_kernel<<<ew((long long)NR * C), 256, 0, stream>>>(accb, ym16, T, NR * C);
    float* tmpo = (float*)walloc((size_t)NR * C * 4);
    gemm(ym16, w_outw, nullptr, tmpo, NR, 256, 256);
    resadd_kernel<<<ew((long long)NR * C), 256, 0, stream>>>(xcur, tmpo, nullptr, C,
                                                             (long long)NR * C);
  }

  // ======================= Phase 4: RevRes ================================
  {
    woff = scratch0;
    h16* hn = (h16*)walloc((size_t)NR * 128 * 2);
    float* tmp = (float*)walloc((size_t)NR * 128 * 4);
    ln(xcur, 256, 128, rv_fn, nullptr, hn, 128, NR);         // ln(x2)
    gemm(hn, w_fw, nullptr, tmp, NR, 128, 128);
    add_strided_kernel<<<ew((long long)NR * 128), 256, 0, stream>>>(
        xcur, 256, 0, tmp, 128, NR * 128);                   // y1 = x1 + .
    ln(xcur, 256, 0, rv_gn, nullptr, hn, 128, NR);           // ln(y1)
    gemm(hn, w_gw, nullptr, tmp, NR, 128, 128);
    add_strided_kernel<<<ew((long long)NR * 128), 256, 0, stream>>>(
        xcur, 256, 128, tmp, 128, NR * 128);                 // y2 = x2 + .
  }

  // ======================= Phase 5: Saliency prune ========================
  {
    woff = scratch0;
    float* xn = (float*)walloc((size_t)NR * C * 4);
    h16* xn16 = (h16*)walloc((size_t)NR * C * 2);
    ln(xcur, C, 0, sl_norm, xn, xn16, C, NR);
    float* hpre = (float*)walloc((size_t)NR * 64 * 4);
    gemm(xn16, w_w1, sl_b1, hpre, NR, 64, 256);
    h16* h16b = (h16*)walloc((size_t)NR * 64 * 2);
    silu_f16_kernel<<<ew((long long)NR * 64), 256, 0, stream>>>(hpre, h16b, NR * 64);
    float* spre = (float*)walloc((size_t)NR * C * 4);
    gemm(h16b, w_w2, sl_b2, spre, NR, 256, 64);
    float* sv = (float*)walloc((size_t)NR * C * 4);
    sigmoid_kernel<<<ew((long long)NR * C), 256, 0, stream>>>(spre, sv, NR * C);
    topk_residual_kernel<<<NR, 256, 0, stream>>>(xcur, xn, sv, sl_gamma,
                                                 (float*)d_out, 205);
  }
}
